// AdaptiveDifferentialAttention_70738111365530
// MI455X (gfx1250) — matrix operand order repaired and checked by reading, not compiled, not run on hardware
//
#include <hip/hip_runtime.h>

// ---------------------------------------------------------------------------
// AdaptiveDifferentialAttention for MI455X (gfx1250, wave32, WMMA bf16)
// Flash-style: attention maps never touch HBM. All global traffic is b128.
// ---------------------------------------------------------------------------

typedef __bf16 bf16_t;
typedef __bf16 v16bf __attribute__((ext_vector_type(16)));
typedef __bf16 v8bf  __attribute__((ext_vector_type(8)));
typedef float  v8f   __attribute__((ext_vector_type(8)));

constexpr int Bc  = 4;
constexpr int Lc  = 1024;
constexpr int Dc  = 512;
constexpr int Hc  = 8;
constexpr int HDc = 64;
constexpr int NW  = Lc / 4;      // 256 windows (ws=4) per (b,h)
constexpr float SCALE = 0.125f;  // 1/sqrt(64)

#if __has_builtin(__builtin_amdgcn_global_load_async_to_lds_b128)
#define HAVE_ASYNC 1
// exact param types per hipcc diagnostic: (int4 as1*, int4 as3*, imm, imm)
typedef int v4i_g __attribute__((vector_size(16)));
typedef __attribute__((address_space(1))) v4i_g* gptr4_t;
typedef __attribute__((address_space(3))) v4i_g* lptr4_t;
#else
#define HAVE_ASYNC 0
#endif

template <int N> __device__ __forceinline__ void wait_async() {
#if __has_builtin(__builtin_amdgcn_s_wait_asynccnt)
    __builtin_amdgcn_s_wait_asynccnt((unsigned short)N);
#else
    asm volatile("s_wait_asynccnt %0" ::"n"(N) : "memory");
#endif
}

union Frag16 { v16bf v; v8bf h[2]; };
union F8 { float4 q[2]; float f[8]; };

// A fragment: lane half h owns K run [16h..16h+15] (matches B)
// -> two contiguous 16B loads.
__device__ __forceinline__ v16bf load_a_bf16(const bf16_t* row, int half) {
    Frag16 f;
    f.h[0] = *(const v8bf*)(row + half * 16);
    f.h[1] = *(const v8bf*)(row + half * 16 + 8);
    return f.v;
}
__device__ __forceinline__ v16bf load_a_f32(const float* row, int half) {
    F8 lo, hi;
    lo.q[0] = *(const float4*)(row + half * 16);
    lo.q[1] = *(const float4*)(row + half * 16 + 4);
    hi.q[0] = *(const float4*)(row + half * 16 + 8);
    hi.q[1] = *(const float4*)(row + half * 16 + 12);
    Frag16 f;
#pragma unroll
    for (int e = 0; e < 8; ++e) {
        f.v[e]     = (bf16_t)lo.f[e];
        f.v[8 + e] = (bf16_t)hi.f[e];
    }
    return f.v;
}
// B fragment: lane = column, element e -> K = 16*half + e: contiguous 16.
__device__ __forceinline__ v16bf load_c16(const bf16_t* p) {
    Frag16 f;
    f.h[0] = *(const v8bf*)(p);
    f.h[1] = *(const v8bf*)(p + 8);
    return f.v;
}

// ---------------------------------------------------------------------------
// Kernel 0: weight prep — W fp32 [K,N] -> Wt bf16 [N,K] (one-time, 0.5MB ea)
// ---------------------------------------------------------------------------
__global__ __launch_bounds__(256) void wprep(const float* __restrict__ W,
                                             bf16_t* __restrict__ Wt) {
    __shared__ float tile[32][33];
    const int n0 = blockIdx.x * 32, k0 = blockIdx.y * 32;
    for (int r = threadIdx.y; r < 32; r += 8)
        tile[r][threadIdx.x] = W[(size_t)(k0 + r) * Dc + n0 + threadIdx.x];
    __syncthreads();
    for (int r = threadIdx.y; r < 32; r += 8)
        Wt[(size_t)(n0 + r) * Dc + k0 + threadIdx.x] = (bf16_t)tile[threadIdx.x][r];
}

// ---------------------------------------------------------------------------
// Kernel 1: Y = X @ W.  X fp32 [4096,512], Wt bf16 [N,K].
// vmode=0 -> bf16 out [B,H,L,HD] (LDS-transposed b128 row stores)
// vmode=1 -> bf16 out [B,H,HD,L] (direct packed b128 stores, for V)
// ---------------------------------------------------------------------------
__global__ __launch_bounds__(128) void proj_gemm_bf16(
    const float* __restrict__ X, const bf16_t* __restrict__ Wt,
    bf16_t* __restrict__ Out, int vmode)
{
    __shared__ bf16_t sbuf[4][16][16];
    const int wid  = threadIdx.x >> 5;
    const int tile = blockIdx.x * 4 + wid;      // 8192 tiles
    const int mt   = tile >> 5;
    const int nt   = tile & 31;
    const int lane = threadIdx.x & 31;
    const int half = lane >> 4, lr = lane & 15;

    const float*  arow = X  + (size_t)(mt * 16 + lr) * Dc;
    const bf16_t* brow = Wt + (size_t)(nt * 16 + lr) * Dc;
    v8f acc = {};
    for (int k0 = 0; k0 < Dc; k0 += 32) {
        __builtin_prefetch(arow + k0 + 64, 0, 3);
        v16bf a = load_a_f32(arow + k0, half);
        v16bf b = load_c16(brow + k0 + half * 16);
        acc = __builtin_amdgcn_wmma_f32_16x16x32_bf16(false, a, false, b,
                                                      (short)0, acc, false, false);
    }
    if (vmode == 0) {
        // stage C/D layout -> row-major rows, store 8 contiguous bf16 / lane
#pragma unroll
        for (int i = 0; i < 8; ++i)
            sbuf[wid][half * 8 + i][lr] = (bf16_t)acc[i];
        v8bf r = *(const v8bf*)&sbuf[wid][lr][half * 8];
        const int n0 = nt * 16 + half * 8, h = n0 >> 6, cd = n0 & 63;
        const int m  = mt * 16 + lr, bb = m >> 10, l = m & (Lc - 1);
        *(v8bf*)&Out[((size_t)(bb * Hc + h) * Lc + l) * HDc + cd] = r;
    } else {
        // V^T layout: element i walks 8 consecutive tokens -> one b128 store
        v8bf r;
#pragma unroll
        for (int i = 0; i < 8; ++i) r[i] = (bf16_t)acc[i];
        const int n = nt * 16 + lr, h = n >> 6, cd = n & 63;
        const int m0 = mt * 16 + half * 8, bb = m0 >> 10, l0 = m0 & (Lc - 1);
        *(v8bf*)&Out[((size_t)(bb * Hc + h) * HDc + cd) * Lc + l0] = r;
    }
}

// ---------------------------------------------------------------------------
// Kernel 2: per-row softmax statistics of both attention maps (no attn in HBM)
// Online accumulators per (row, lane): m=max logit, mn=min logit,
// z=sum exp(s-m), w=sum exp(2(s-m)).  exp(2x)=exp(x)^2 -> 2 exps per update.
// ---------------------------------------------------------------------------
__global__ __launch_bounds__(128) void attn_stats(
    const bf16_t* __restrict__ Q1, const bf16_t* __restrict__ K1,
    const bf16_t* __restrict__ Q2, const bf16_t* __restrict__ K2,
    float* __restrict__ rs1, float* __restrict__ rs2,   // [B*H*L][2] = {m, Z}
    float* __restrict__ gate_in)                        // [B*H*NW][8]
{
    const int wid = threadIdx.x >> 5;
    const int idx = blockIdx.x * 4 + wid;       // 2048 waves
    const int b   = idx >> 9;
    const int h   = (idx >> 6) & 7;
    const int qt  = idx & 63;
    const int lane = threadIdx.x & 31;
    const int half = lane >> 4, lr = lane & 15;
    const int bhBase = (b * Hc + h) * Lc;

    const bf16_t* q1p = Q1 + (size_t)(bhBase + qt * 16 + lr) * HDc;
    const bf16_t* q2p = Q2 + (size_t)(bhBase + qt * 16 + lr) * HDc;
    const v16bf qa10 = load_a_bf16(q1p, half),      qa11 = load_a_bf16(q1p + 32, half);
    const v16bf qa20 = load_a_bf16(q2p, half),      qa21 = load_a_bf16(q2p + 32, half);

    float m1[8], mn1[8], z1[8], w1[8], m2[8], mn2[8], z2[8], w2[8];
#pragma unroll
    for (int i = 0; i < 8; ++i) {
        m1[i] = m2[i] = -3.0e38f;  mn1[i] = mn2[i] = 3.0e38f;
        z1[i] = z2[i] = 0.f;       w1[i] = w2[i] = 0.f;
    }

    const bf16_t* k1base = K1 + (size_t)bhBase * HDc;
    const bf16_t* k2base = K2 + (size_t)bhBase * HDc;
    for (int kt = 0; kt < Lc / 16; ++kt) {
        const bf16_t* k1p = k1base + (size_t)(kt * 16 + lr) * HDc;
        const bf16_t* k2p = k2base + (size_t)(kt * 16 + lr) * HDc;
        v16bf b10 = load_c16(k1p + half * 16), b11 = load_c16(k1p + 32 + half * 16);
        v16bf b20 = load_c16(k2p + half * 16), b21 = load_c16(k2p + 32 + half * 16);
        v8f s1 = {}, s2 = {};
        s1 = __builtin_amdgcn_wmma_f32_16x16x32_bf16(false, qa10, false, b10, (short)0, s1, false, false);
        s1 = __builtin_amdgcn_wmma_f32_16x16x32_bf16(false, qa11, false, b11, (short)0, s1, false, false);
        s2 = __builtin_amdgcn_wmma_f32_16x16x32_bf16(false, qa20, false, b20, (short)0, s2, false, false);
        s2 = __builtin_amdgcn_wmma_f32_16x16x32_bf16(false, qa21, false, b21, (short)0, s2, false, false);
#pragma unroll
        for (int i = 0; i < 8; ++i) {
            float s  = s1[i] * SCALE;
            float nm = fmaxf(m1[i], s);
            float r  = __expf(m1[i] - nm);   // exp(2x) = exp(x)^2
            float e  = __expf(s - nm);
            z1[i] = z1[i] * r + e;
            w1[i] = w1[i] * r * r + e * e;
            m1[i] = nm;  mn1[i] = fminf(mn1[i], s);
            s  = s2[i] * SCALE;
            nm = fmaxf(m2[i], s);
            r  = __expf(m2[i] - nm);
            e  = __expf(s - nm);
            z2[i] = z2[i] * r + e;
            w2[i] = w2[i] * r * r + e * e;
            m2[i] = nm;  mn2[i] = fminf(mn2[i], s);
        }
    }
    // merge across the 16 lanes of each half (xor masks < 16)
#pragma unroll
    for (int off = 1; off < 16; off <<= 1) {
#pragma unroll
        for (int i = 0; i < 8; ++i) {
            float mo = __shfl_xor(m1[i], off, 32);
            float zo = __shfl_xor(z1[i], off, 32);
            float wo = __shfl_xor(w1[i], off, 32);
            float nm = fmaxf(m1[i], mo);
            float ra = __expf(m1[i] - nm), rb = __expf(mo - nm);
            z1[i] = z1[i] * ra + zo * rb;
            w1[i] = w1[i] * ra * ra + wo * rb * rb;
            m1[i] = nm;
            mn1[i] = fminf(mn1[i], __shfl_xor(mn1[i], off, 32));
            mo = __shfl_xor(m2[i], off, 32);
            zo = __shfl_xor(z2[i], off, 32);
            wo = __shfl_xor(w2[i], off, 32);
            nm = fmaxf(m2[i], mo);
            ra = __expf(m2[i] - nm);  rb = __expf(mo - nm);
            z2[i] = z2[i] * ra + zo * rb;
            w2[i] = w2[i] * ra * ra + wo * rb * rb;
            m2[i] = nm;
            mn2[i] = fminf(mn2[i], __shfl_xor(mn2[i], off, 32));
        }
    }
    if (lr == 0) {
        float st1[8], mx1[8], mi1[8], st2[8], mx2[8], mi2[8];
#pragma unroll
        for (int i = 0; i < 8; ++i) {
            int q = qt * 16 + half * 8 + i;
            rs1[(size_t)(bhBase + q) * 2 + 0] = m1[i];
            rs1[(size_t)(bhBase + q) * 2 + 1] = z1[i];
            rs2[(size_t)(bhBase + q) * 2 + 0] = m2[i];
            rs2[(size_t)(bhBase + q) * 2 + 1] = z2[i];
            // p-statistics: mean p = 1/L (exact), sum p^2 = w/z^2,
            // max p = 1/z, min p = exp(mn-m)/z
            float sq = w1[i] / (z1[i] * z1[i]);
            st1[i] = sqrtf(fmaxf((sq - 1.0f / Lc) / (Lc - 1), 0.f));
            mx1[i] = 1.0f / z1[i];
            mi1[i] = __expf(mn1[i] - m1[i]) / z1[i];
            sq = w2[i] / (z2[i] * z2[i]);
            st2[i] = sqrtf(fmaxf((sq - 1.0f / Lc) / (Lc - 1), 0.f));
            mx2[i] = 1.0f / z2[i];
            mi2[i] = __expf(mn2[i] - m2[i]) / z2[i];
        }
#pragma unroll
        for (int wl = 0; wl < 2; ++wl) {
            float a1 = 0, a2 = 0, a3 = 0, c1 = 0, c2 = 0, c3 = 0;
#pragma unroll
            for (int i = 0; i < 4; ++i) {
                a1 += st1[wl * 4 + i]; a2 += mx1[wl * 4 + i]; a3 += mi1[wl * 4 + i];
                c1 += st2[wl * 4 + i]; c2 += mx2[wl * 4 + i]; c3 += mi2[wl * 4 + i];
            }
            int w = qt * 4 + half * 2 + wl;
            float* g = gate_in + (size_t)((b * Hc + h) * NW + w) * 8;
            g[0] = 1.0f / Lc; g[1] = a1 * 0.25f; g[2] = a2 * 0.25f; g[3] = a3 * 0.25f;
            g[4] = 1.0f / Lc; g[5] = c1 * 0.25f; g[6] = c2 * 0.25f; g[7] = c3 * 0.25f;
        }
    }
}

// ---------------------------------------------------------------------------
// Kernel 3: gate MLP 8 -> 32 -> 16 -> 1 (sigmoid)
// ---------------------------------------------------------------------------
__global__ __launch_bounds__(256) void gate_mlp(
    const float* __restrict__ gin,
    const float* __restrict__ gw1, const float* __restrict__ gb1,
    const float* __restrict__ gw2, const float* __restrict__ gb2,
    const float* __restrict__ gw3, const float* __restrict__ gb3,
    float* __restrict__ alpha)
{
    int i = blockIdx.x * blockDim.x + threadIdx.x;
    if (i >= Bc * Hc * NW) return;
    const float* g = gin + (size_t)i * 8;
    float h1[32];
#pragma unroll
    for (int j = 0; j < 32; ++j) {
        float s = gb1[j];
#pragma unroll
        for (int k = 0; k < 8; ++k) s += g[k] * gw1[k * 32 + j];
        h1[j] = fmaxf(s, 0.f);
    }
    float h2[16];
#pragma unroll
    for (int j = 0; j < 16; ++j) {
        float s = gb2[j];
#pragma unroll
        for (int k = 0; k < 32; ++k) s += h1[k] * gw2[k * 16 + j];
        h2[j] = fmaxf(s, 0.f);
    }
    float s = gb3[0];
#pragma unroll
    for (int k = 0; k < 16; ++k) s += h2[k] * gw3[k];
    alpha[i] = 1.f / (1.f + __expf(-s));
}

// ---------------------------------------------------------------------------
// Kernel 4: fused flash pass with async double-buffered V^T tiles.
// ---------------------------------------------------------------------------
__device__ __forceinline__ void fetch_vtile(const bf16_t* vtbh, int t,
                                            bf16_t* dst /*[64][32]*/, int lane) {
    const int hd = lane >> 2;            // 0..7
    const int co = (lane & 3) * 8;       // 16B column chunk
#pragma unroll
    for (int j = 0; j < 8; ++j) {
        const bf16_t* src = vtbh + (size_t)(j * 8 + hd) * Lc + t * 32 + co;
        bf16_t* d = dst + (j * 8 + hd) * 32 + co;
#if HAVE_ASYNC
        __builtin_amdgcn_global_load_async_to_lds_b128(
            (gptr4_t)(unsigned long long)(const void*)src,
            (lptr4_t)(unsigned long long)(void*)d, 0, 0);
#else
        *(v8bf*)d = *(const v8bf*)src;
#endif
    }
}

__global__ __launch_bounds__(128) void attn_fused(
    const bf16_t* __restrict__ Q1, const bf16_t* __restrict__ K1,
    const bf16_t* __restrict__ Q2, const bf16_t* __restrict__ K2,
    const bf16_t* __restrict__ Vt /* [B,H,HD,L] */,
    const float* __restrict__ rs1, const float* __restrict__ rs2,
    const float* __restrict__ alpha,
    bf16_t* __restrict__ Out /* [B,L,D] */)
{
    __shared__ bf16_t pbuf[4][16][32];
    __shared__ bf16_t vbuf[4][2][64][32];
    __shared__ bf16_t obuf[4][16][64];
    const int wid = threadIdx.x >> 5;
    const int idx = blockIdx.x * 4 + wid;
    const int b   = idx >> 9;
    const int h   = (idx >> 6) & 7;
    const int qt  = idx & 63;
    const int lane = threadIdx.x & 31;
    const int half = lane >> 4, lr = lane & 15;
    const int bhBase = (b * Hc + h) * Lc;

    const bf16_t* q1p = Q1 + (size_t)(bhBase + qt * 16 + lr) * HDc;
    const bf16_t* q2p = Q2 + (size_t)(bhBase + qt * 16 + lr) * HDc;
    const v16bf qa10 = load_a_bf16(q1p, half),      qa11 = load_a_bf16(q1p + 32, half);
    const v16bf qa20 = load_a_bf16(q2p, half),      qa21 = load_a_bf16(q2p + 32, half);

    float al[8], mm1[8], iz1[8], mm2[8], iz2[8];
#pragma unroll
    for (int i = 0; i < 8; ++i) {
        int q = qt * 16 + half * 8 + i;
        al[i]  = alpha[(size_t)(b * Hc + h) * NW + (q >> 2)];
        mm1[i] = rs1[(size_t)(bhBase + q) * 2 + 0];
        iz1[i] = 1.f / rs1[(size_t)(bhBase + q) * 2 + 1];
        mm2[i] = rs2[(size_t)(bhBase + q) * 2 + 0];
        iz2[i] = 1.f / rs2[(size_t)(bhBase + q) * 2 + 1];
    }

    const bf16_t* k1base = K1 + (size_t)bhBase * HDc;
    const bf16_t* k2base = K2 + (size_t)bhBase * HDc;
    const bf16_t* vtbh   = Vt + (size_t)((b * Hc + h) * HDc) * Lc;

    fetch_vtile(vtbh, 0, &vbuf[wid][0][0][0], lane);
    v8f o[4] = {v8f{}, v8f{}, v8f{}, v8f{}};
    for (int t = 0; t < Lc / 32; ++t) {
        if (t + 1 < Lc / 32)
            fetch_vtile(vtbh, t + 1, &vbuf[wid][(t + 1) & 1][0][0], lane);
#if HAVE_ASYNC
        if (t + 1 < Lc / 32) wait_async<8>(); else wait_async<0>();
#endif
#pragma unroll
        for (int sub = 0; sub < 2; ++sub) {
            const int kt = t * 2 + sub;
            const bf16_t* k1p = k1base + (size_t)(kt * 16 + lr) * HDc;
            const bf16_t* k2p = k2base + (size_t)(kt * 16 + lr) * HDc;
            v16bf b10 = load_c16(k1p + half * 16), b11 = load_c16(k1p + 32 + half * 16);
            v16bf b20 = load_c16(k2p + half * 16), b21 = load_c16(k2p + 32 + half * 16);
            v8f s1 = {}, s2 = {};
            s1 = __builtin_amdgcn_wmma_f32_16x16x32_bf16(false, qa10, false, b10, (short)0, s1, false, false);
            s1 = __builtin_amdgcn_wmma_f32_16x16x32_bf16(false, qa11, false, b11, (short)0, s1, false, false);
            s2 = __builtin_amdgcn_wmma_f32_16x16x32_bf16(false, qa20, false, b20, (short)0, s2, false, false);
            s2 = __builtin_amdgcn_wmma_f32_16x16x32_bf16(false, qa21, false, b21, (short)0, s2, false, false);
#pragma unroll
            for (int i = 0; i < 8; ++i) {
                float p1 = __expf(s1[i] * SCALE - mm1[i]) * iz1[i];
                float p2 = __expf(s2[i] * SCALE - mm2[i]) * iz2[i];
                pbuf[wid][half * 8 + i][sub * 16 + lr] =
                    (bf16_t)(al[i] * p1 + (1.f - al[i]) * p2);
            }
        }
        // reload fused P in A-fragment layout (K = 16*half + e): two contiguous 16B LDS reads
        Frag16 pf;
        pf.h[0] = *(const v8bf*)&pbuf[wid][lr][half * 16];
        pf.h[1] = *(const v8bf*)&pbuf[wid][lr][half * 16 + 8];
        const bf16_t* vt = &vbuf[wid][t & 1][0][0];
#pragma unroll
        for (int nt = 0; nt < 4; ++nt) {
            v16bf vf = load_c16(vt + (nt * 16 + lr) * 32 + half * 16);
            o[nt] = __builtin_amdgcn_wmma_f32_16x16x32_bf16(false, pf.v, false, vf,
                                                            (short)0, o[nt], false, false);
        }
    }
    // stage 16x64 tile, store 64B contiguous per lane
#pragma unroll
    for (int nt = 0; nt < 4; ++nt)
#pragma unroll
        for (int i = 0; i < 8; ++i)
            obuf[wid][half * 8 + i][nt * 16 + lr] = (bf16_t)o[nt][i];
    bf16_t* orow = &Out[(size_t)(b * Lc + qt * 16 + lr) * Dc + h * HDc + half * 32];
#pragma unroll
    for (int j = 0; j < 4; ++j)
        *(v8bf*)(orow + j * 8) = *(const v8bf*)&obuf[wid][lr][half * 32 + j * 8];
}

// ---------------------------------------------------------------------------
// Kernel 5: out = AttnOut(bf16 [4096,512]) @ Wo^T(bf16 [N,K]) -> fp32 d_out
// ---------------------------------------------------------------------------
__global__ __launch_bounds__(128) void out_proj(
    const bf16_t* __restrict__ A, const bf16_t* __restrict__ Wt,
    float* __restrict__ Out)
{
    __shared__ float ob[4][16][16];
    const int wid  = threadIdx.x >> 5;
    const int tile = blockIdx.x * 4 + wid;
    const int mt   = tile >> 5;
    const int nt   = tile & 31;
    const int lane = threadIdx.x & 31;
    const int half = lane >> 4, lr = lane & 15;

    const bf16_t* arow = A  + (size_t)(mt * 16 + lr) * Dc;
    const bf16_t* brow = Wt + (size_t)(nt * 16 + lr) * Dc;
    v8f acc = {};
    for (int k0 = 0; k0 < Dc; k0 += 32) {
        v16bf a = load_a_bf16(arow + k0, half);
        v16bf b = load_c16(brow + k0 + half * 16);
        acc = __builtin_amdgcn_wmma_f32_16x16x32_bf16(false, a, false, b,
                                                      (short)0, acc, false, false);
    }
#pragma unroll
    for (int i = 0; i < 8; ++i) ob[wid][half * 8 + i][lr] = acc[i];
    float* out0 = &Out[(size_t)(mt * 16 + lr) * Dc + nt * 16 + half * 8];
    *(float4*)(out0)     = *(const float4*)&ob[wid][lr][half * 8];
    *(float4*)(out0 + 4) = *(const float4*)&ob[wid][lr][half * 8 + 4];
}

// ---------------------------------------------------------------------------
extern "C" void kernel_launch(void* const* d_in, const int* in_sizes, int n_in,
                              void* d_out, int out_size, void* d_ws, size_t ws_size,
                              hipStream_t stream) {
    const float* x   = (const float*)d_in[0];
    const float* Wf[6] = {(const float*)d_in[1], (const float*)d_in[2],
                          (const float*)d_in[3], (const float*)d_in[4],
                          (const float*)d_in[5], (const float*)d_in[6]};
    const float* gw1 = (const float*)d_in[7];
    const float* gb1 = (const float*)d_in[8];
    const float* gw2 = (const float*)d_in[9];
    const float* gb2 = (const float*)d_in[10];
    const float* gw3 = (const float*)d_in[11];
    const float* gb3 = (const float*)d_in[12];

    char* ws = (char*)d_ws;
    constexpr size_t BY_QKV = (size_t)Bc * Hc * Lc * HDc * sizeof(bf16_t); // 4MB
    bf16_t* q1   = (bf16_t*)(ws + 0 * BY_QKV);
    bf16_t* k1   = (bf16_t*)(ws + 1 * BY_QKV);
    bf16_t* q2   = (bf16_t*)(ws + 2 * BY_QKV);
    bf16_t* k2   = (bf16_t*)(ws + 3 * BY_QKV);
    bf16_t* vt   = (bf16_t*)(ws + 4 * BY_QKV);
    bf16_t* aout = (bf16_t*)(ws + 5 * BY_QKV);
    float*  rs1   = (float*)(ws + 6 * BY_QKV);
    float*  rs2   = rs1 + (size_t)Bc * Hc * Lc * 2;
    float*  gin   = rs2 + (size_t)Bc * Hc * Lc * 2;
    float*  alpha = gin + (size_t)Bc * Hc * NW * 8;
    bf16_t* wt0   = (bf16_t*)(alpha + (size_t)Bc * Hc * NW);
    bf16_t* wt[6];
    for (int i = 0; i < 6; ++i) wt[i] = wt0 + (size_t)i * Dc * Dc;

    dim3 tb(32, 8);
    for (int i = 0; i < 6; ++i)
        wprep<<<dim3(16, 16), tb, 0, stream>>>(Wf[i], wt[i]);

    dim3 blk(128);
    proj_gemm_bf16<<<2048, blk, 0, stream>>>(x, wt[0], q1, 0);  // Wq1
    proj_gemm_bf16<<<2048, blk, 0, stream>>>(x, wt[1], k1, 0);  // Wk1
    proj_gemm_bf16<<<2048, blk, 0, stream>>>(x, wt[2], q2, 0);  // Wq2
    proj_gemm_bf16<<<2048, blk, 0, stream>>>(x, wt[3], k2, 0);  // Wk2
    proj_gemm_bf16<<<2048, blk, 0, stream>>>(x, wt[4], vt, 1);  // Wv -> V^T

    attn_stats<<<512, blk, 0, stream>>>(q1, k1, q2, k2, rs1, rs2, gin);
    gate_mlp<<<32, 256, 0, stream>>>(gin, gw1, gb1, gw2, gb2, gw3, gb3, alpha);
    attn_fused<<<512, blk, 0, stream>>>(q1, k1, q2, k2, vt, rs1, rs2, alpha, aout);
    out_proj<<<2048, blk, 0, stream>>>(aout, wt[5], (float*)d_out);
}
